// GNNEncoder_17351667876348
// MI455X (gfx1250) — matrix operand order repaired and checked by reading, not compiled, not run on hardware
//
#include <hip/hip_runtime.h>
#include <hip/hip_bf16.h>

typedef __attribute__((ext_vector_type(16))) _Float16 v16h;
typedef __attribute__((ext_vector_type(8)))  _Float16 v8h;
typedef __attribute__((ext_vector_type(8)))  float    v8f;

// ---------------------------------------------------------------------------
// A-fragment loader: 16 f32 -> f16, following CDNA5 16-bit A 16x32 layout.
// Lane (row = lane&15, half = lane>>4):
//   a[j], j=0..7  : K = 8*half + j
//   a[j], j=8..15 : K = 16 + 8*half + (j-8)
// p points at the start of a 32-float row.
// ---------------------------------------------------------------------------
__device__ __forceinline__ v16h load_a_row_f32(const float* __restrict__ p, int half) {
    const float4* q0 = (const float4*)(p + half * 8);
    const float4* q1 = (const float4*)(p + 16 + half * 8);
    float4 f0 = q0[0], f1 = q0[1];
    float4 f2 = q1[0], f3 = q1[1];
    v16h a;
    a[0]  = (_Float16)f0.x; a[1]  = (_Float16)f0.y; a[2]  = (_Float16)f0.z; a[3]  = (_Float16)f0.w;
    a[4]  = (_Float16)f1.x; a[5]  = (_Float16)f1.y; a[6]  = (_Float16)f1.z; a[7]  = (_Float16)f1.w;
    a[8]  = (_Float16)f2.x; a[9]  = (_Float16)f2.y; a[10] = (_Float16)f2.z; a[11] = (_Float16)f2.w;
    a[12] = (_Float16)f3.x; a[13] = (_Float16)f3.y; a[14] = (_Float16)f3.z; a[15] = (_Float16)f3.w;
    return a;
}

// B-fragment: weights stored transposed in f16: wt[col*stride + k].
// Lane (col = lane&15, half = lane>>4) holds K = 8*half + j for j=0..7,
// and K = 16 + 8*half + (j-8) for j=8..15 -> two b128 loads.
__device__ __forceinline__ v16h load_b(const _Float16* __restrict__ wt,
                                       int col, int stride, int kofs, int half) {
    const _Float16* p = wt + (size_t)col * stride + kofs + half * 8;
    v8h lo = *(const v8h*)(p);
    v8h hi = *(const v8h*)(p + 16);
    return __builtin_shufflevector(lo, hi, 0, 1, 2, 3, 4, 5, 6, 7, 8, 9, 10, 11, 12, 13, 14, 15);
}

// ---------------------------------------------------------------------------
// Weight prep: transpose + f32->f16 (edge_w1 padded 67 -> 96 rows)
// ---------------------------------------------------------------------------
__global__ void prep_weights(const float* __restrict__ gw1, const float* __restrict__ gw2,
                             const float* __restrict__ ew1, const float* __restrict__ ew2,
                             _Float16* __restrict__ gw1t, _Float16* __restrict__ gw2t,
                             _Float16* __restrict__ ew1t, _Float16* __restrict__ ew2t) {
    for (int i = threadIdx.x; i < 32 * 32; i += blockDim.x) {
        int col = i >> 5, k = i & 31;
        gw1t[col * 32 + k] = (_Float16)gw1[k * 32 + col];
        gw2t[col * 32 + k] = (_Float16)gw2[k * 32 + col];
        ew2t[col * 32 + k] = (_Float16)ew2[k * 32 + col];
    }
    for (int i = threadIdx.x; i < 32 * 96; i += blockDim.x) {
        int col = i / 96, k = i % 96;
        ew1t[col * 96 + k] = (k < 67) ? (_Float16)ew1[k * 32 + col] : (_Float16)0.0f;
    }
}

// ---------------------------------------------------------------------------
// Degree / normalization  (dis[] = D^{-1/2}, 400 KB -> L2 resident)
// ---------------------------------------------------------------------------
__global__ void deg_init(float* __restrict__ deg, int n) {
    int i = blockIdx.x * blockDim.x + threadIdx.x;
    if (i < n) deg[i] = 1.0f;   // self-loop contributes 1
}
__global__ void deg_count(const int* __restrict__ dst, float* __restrict__ deg, int nE) {
    int e = blockIdx.x * blockDim.x + threadIdx.x;
    if (e < nE) atomicAdd(&deg[dst[e]], 1.0f);
}
__global__ void deg_rsqrt(float* __restrict__ deg, int n) {
    int i = blockIdx.x * blockDim.x + threadIdx.x;
    if (i < n) deg[i] = rsqrtf(deg[i]);   // deg >= 1 always
}

// ---------------------------------------------------------------------------
// Node encoder MLP: h = relu(x@W1+b1)@W2+b2   (tiny, one thread per node)
// ---------------------------------------------------------------------------
__global__ void node_encoder(const float* __restrict__ x,
                             const float* __restrict__ w1, const float* __restrict__ b1,
                             const float* __restrict__ w2, const float* __restrict__ b2,
                             float* __restrict__ h, int n) {
    int i = blockIdx.x * blockDim.x + threadIdx.x;
    if (i >= n) return;
    float x0 = x[2 * i], x1 = x[2 * i + 1];
    float hid[32];
#pragma unroll
    for (int j = 0; j < 32; ++j) {
        float v = fmaf(x0, w1[j], fmaf(x1, w1[32 + j], b1[j]));
        hid[j] = v > 0.0f ? v : 0.0f;
    }
#pragma unroll
    for (int j = 0; j < 32; ++j) {
        float acc = b2[j];
#pragma unroll
        for (int k = 0; k < 32; ++k) acc = fmaf(hid[k], w2[k * 32 + j], acc);
        h[(size_t)i * 32 + j] = acc;
    }
}

// ---------------------------------------------------------------------------
// GCN dense transform: hw = h @ W via WMMA. One wave per 16-node tile,
// two 16x16 col tiles, single K=32 step each. Uniform fast path for full
// tiles (no per-element branches); guarded path only for the tail tile.
// ---------------------------------------------------------------------------
__global__ void gcn_transform(const float* __restrict__ h, const _Float16* __restrict__ wt,
                              float* __restrict__ hw, int nTiles, int nNodes) {
    int wave = (int)((blockIdx.x * blockDim.x + threadIdx.x) >> 5);
    if (wave >= nTiles) return;                 // uniform per wave
    int lane = threadIdx.x & 31;
    int row  = lane & 15, half = lane >> 4, col = row;
    int node = wave * 16 + row;
    if (node >= nNodes) node = nNodes - 1;      // clamp loads (stores guarded)

    v16h a  = load_a_row_f32(h + (size_t)node * 32, half);
    v16h b0 = load_b(wt, col,      32, 0, half);
    v16h b1 = load_b(wt, col + 16, 32, 0, half);
    v8f c0 = {}; v8f c1 = {};
    c0 = __builtin_amdgcn_wmma_f32_16x16x32_f16(false, a, false, b0, (short)0, c0, false, false);
    c1 = __builtin_amdgcn_wmma_f32_16x16x32_f16(false, a, false, b1, (short)0, c1, false, false);

    int base = wave * 16 + half * 8;            // D layout: M = r + 8*half, N = col
    if (wave * 16 + 16 <= nNodes) {             // uniform: full tile, straight-line stores
#pragma unroll
        for (int r = 0; r < 8; ++r) {
            float* o = hw + (size_t)(base + r) * 32 + col;
            o[0]  = c0[r];
            o[16] = c1[r];
        }
    } else {                                    // tail tile only
#pragma unroll
        for (int r = 0; r < 8; ++r) {
            if (base + r < nNodes) {
                float* o = hw + (size_t)(base + r) * 32 + col;
                o[0]  = c0[r];
                o[16] = c1[r];
            }
        }
    }
}

// acc[n][c] = hw[n][c] * dis[n]^2   (self-loop term initializes the segment sum)
__global__ void gcn_init_self(const float* __restrict__ hw, const float* __restrict__ dis,
                              float* __restrict__ acc, int total) {
    int i = blockIdx.x * blockDim.x + threadIdx.x;
    if (i >= total) return;
    float s = dis[i >> 5];
    acc[i] = hw[i] * s * s;
}

// scatter: acc[dst] += hw[src] * dis[src]*dis[dst]   (8 threads/edge, float4 each;
// dis[] is L2-resident so the two gathers beat streaming a precomputed norm[E])
__global__ void gcn_scatter(const float* __restrict__ hw, const int* __restrict__ src,
                            const int* __restrict__ dst, const float* __restrict__ dis,
                            float* __restrict__ acc, int nE) {
    int idx = blockIdx.x * blockDim.x + threadIdx.x;
    int e = idx >> 3;
    if (e >= nE) return;
    int c = (idx & 7) << 2;
    int s = src[e], d = dst[e];
    float nm = dis[s] * dis[d];
    float4 v = *(const float4*)(hw + (size_t)s * 32 + c);
    float* o = acc + (size_t)d * 32 + c;
    atomicAdd(o + 0, v.x * nm);
    atomicAdd(o + 1, v.y * nm);
    atomicAdd(o + 2, v.z * nm);
    atomicAdd(o + 3, v.w * nm);
}

__global__ void bias_relu(const float* __restrict__ acc, const float* __restrict__ b,
                          float* __restrict__ out, int total) {
    int i = blockIdx.x * blockDim.x + threadIdx.x;
    if (i >= total) return;
    float v = acc[i] + b[i & 31];
    out[i] = v > 0.0f ? v : 0.0f;
}

// ---------------------------------------------------------------------------
// Edge MLP: e = relu([h_src|h_dst|attr] @ W1 + b1) @ W2 + b2
// One wave per 16-edge tile. Layer 1: 3 K-chunks x 2 col tiles = 6 WMMAs.
// Layer-1 output is re-laid-out A-style via a per-wave LDS tile, then
// layer 2: 2 WMMAs.
// ---------------------------------------------------------------------------
__global__ void edge_mlp(const float* __restrict__ h,
                         const int* __restrict__ src, const int* __restrict__ dst,
                         const float* __restrict__ attr,
                         const _Float16* __restrict__ w1t, const float* __restrict__ b1,
                         const _Float16* __restrict__ w2t, const float* __restrict__ b2,
                         float* __restrict__ eout, int nTiles) {
    __shared__ _Float16 stage[8][16 * 32];
    int wib  = threadIdx.x >> 5;
    int tile = blockIdx.x * 8 + wib;
    bool active = (tile < nTiles);              // uniform per wave
    int lane = threadIdx.x & 31;
    int row  = lane & 15, half = lane >> 4, col = row;
    _Float16* st = stage[wib];

    if (active) {
        int e = tile * 16 + row;
        int s = src[e], d = dst[e];
        v16h a0 = load_a_row_f32(h + (size_t)s * 32, half);   // K chunk 0: h[src]
        v16h a1 = load_a_row_f32(h + (size_t)d * 32, half);   // K chunk 1: h[dst]
        v16h a2;                                              // K chunk 2: attr (padded)
#pragma unroll
        for (int j = 0; j < 16; ++j) a2[j] = (_Float16)0.0f;
        if (half == 0) {                                      // K'=0..2 live in lanes 0-15
            a2[0] = (_Float16)attr[(size_t)e * 3 + 0];
            a2[1] = (_Float16)attr[(size_t)e * 3 + 1];
            a2[2] = (_Float16)attr[(size_t)e * 3 + 2];
        }
        v8f c0 = {}; v8f c1 = {};
        c0 = __builtin_amdgcn_wmma_f32_16x16x32_f16(false, a0, false, load_b(w1t, col,      96,  0, half), (short)0, c0, false, false);
        c0 = __builtin_amdgcn_wmma_f32_16x16x32_f16(false, a1, false, load_b(w1t, col,      96, 32, half), (short)0, c0, false, false);
        c0 = __builtin_amdgcn_wmma_f32_16x16x32_f16(false, a2, false, load_b(w1t, col,      96, 64, half), (short)0, c0, false, false);
        c1 = __builtin_amdgcn_wmma_f32_16x16x32_f16(false, a0, false, load_b(w1t, col + 16, 96,  0, half), (short)0, c1, false, false);
        c1 = __builtin_amdgcn_wmma_f32_16x16x32_f16(false, a1, false, load_b(w1t, col + 16, 96, 32, half), (short)0, c1, false, false);
        c1 = __builtin_amdgcn_wmma_f32_16x16x32_f16(false, a2, false, load_b(w1t, col + 16, 96, 64, half), (short)0, c1, false, false);

        float bc0 = b1[col], bc1 = b1[col + 16];
#pragma unroll
        for (int r = 0; r < 8; ++r) {
            int M = r + half * 8;
            float v0 = c0[r] + bc0; v0 = v0 > 0.0f ? v0 : 0.0f;
            float v1 = c1[r] + bc1; v1 = v1 > 0.0f ? v1 : 0.0f;
            st[M * 32 + col]      = (_Float16)v0;
            st[M * 32 + col + 16] = (_Float16)v1;
        }
    }
    __syncthreads();
    if (active) {
        // Re-read layer-1 activations in A layout.
        v8h lo = *(const v8h*)(st + row * 32 + half * 8);
        v8h hi = *(const v8h*)(st + row * 32 + 16 + half * 8);
        v16h a;
#pragma unroll
        for (int j = 0; j < 8; ++j) { a[j] = lo[j]; a[j + 8] = hi[j]; }
        v8f d0 = {}; v8f d1 = {};
        d0 = __builtin_amdgcn_wmma_f32_16x16x32_f16(false, a, false, load_b(w2t, col,      32, 0, half), (short)0, d0, false, false);
        d1 = __builtin_amdgcn_wmma_f32_16x16x32_f16(false, a, false, load_b(w2t, col + 16, 32, 0, half), (short)0, d1, false, false);
        float bc0 = b2[col], bc1 = b2[col + 16];
#pragma unroll
        for (int r = 0; r < 8; ++r) {
            float* o = eout + (size_t)(tile * 16 + r + half * 8) * 32 + col;
            o[0]  = d0[r] + bc0;
            o[16] = d1[r] + bc1;
        }
    }
}

// ---------------------------------------------------------------------------
extern "C" void kernel_launch(void* const* d_in, const int* in_sizes, int n_in,
                              void* d_out, int out_size, void* d_ws, size_t ws_size,
                              hipStream_t stream) {
    const float* x    = (const float*)d_in[0];
    const int*   eidx = (const int*)d_in[1];
    const float* eatt = (const float*)d_in[2];
    const float* nw1  = (const float*)d_in[3];  const float* nb1 = (const float*)d_in[4];
    const float* nw2  = (const float*)d_in[5];  const float* nb2 = (const float*)d_in[6];
    const float* gw1  = (const float*)d_in[7];  const float* gb1 = (const float*)d_in[8];
    const float* gw2  = (const float*)d_in[9];  const float* gb2 = (const float*)d_in[10];
    const float* ew1  = (const float*)d_in[11]; const float* eb1 = (const float*)d_in[12];
    const float* ew2  = (const float*)d_in[13]; const float* eb2 = (const float*)d_in[14];

    const int N = in_sizes[0] / 2;
    const int E = in_sizes[2] / 3;
    const int* src = eidx;
    const int* dst = eidx + E;

    float* out_h = (float*)d_out;            // [N,32]
    float* out_e = out_h + (size_t)N * 32;   // [E,32]

    char* ws = (char*)d_ws;
    size_t off = 0;
    auto alloc = [&](size_t bytes) -> char* {
        char* p = ws + off;
        off += (bytes + 255) & ~(size_t)255;
        return p;
    };
    float*     dis  = (float*)alloc((size_t)N * 4);
    float*     hA   = (float*)alloc((size_t)N * 32 * 4);
    float*     hw   = (float*)alloc((size_t)N * 32 * 4);
    float*     acc  = (float*)alloc((size_t)N * 32 * 4);
    _Float16*  gw1t = (_Float16*)alloc(32 * 32 * 2);
    _Float16*  gw2t = (_Float16*)alloc(32 * 32 * 2);
    _Float16*  ew1t = (_Float16*)alloc(32 * 96 * 2);
    _Float16*  ew2t = (_Float16*)alloc(32 * 32 * 2);
    (void)ws_size; (void)n_in; (void)out_size;

    const int B = 256;
    auto cdiv = [](int a, int b) { return (a + b - 1) / b; };

    prep_weights<<<1, B, 0, stream>>>(gw1, gw2, ew1, ew2, gw1t, gw2t, ew1t, ew2t);
    deg_init   <<<cdiv(N, B), B, 0, stream>>>(dis, N);
    deg_count  <<<cdiv(E, B), B, 0, stream>>>(dst, dis, E);
    deg_rsqrt  <<<cdiv(N, B), B, 0, stream>>>(dis, N);
    node_encoder<<<cdiv(N, B), B, 0, stream>>>(x, nw1, nb1, nw2, nb2, hA, N);

    const int nTileN = cdiv(N, 16);
    const int NH = N * 32;
    // GCN layer 1
    gcn_transform<<<cdiv(nTileN, 8), B, 0, stream>>>(hA, gw1t, hw, nTileN, N);
    gcn_init_self<<<cdiv(NH, B), B, 0, stream>>>(hw, dis, acc, NH);
    gcn_scatter  <<<cdiv(E * 8, B), B, 0, stream>>>(hw, src, dst, dis, acc, E);
    bias_relu    <<<cdiv(NH, B), B, 0, stream>>>(acc, gb1, hA, NH);
    // GCN layer 2 -> final h straight into d_out
    gcn_transform<<<cdiv(nTileN, 8), B, 0, stream>>>(hA, gw2t, hw, nTileN, N);
    gcn_init_self<<<cdiv(NH, B), B, 0, stream>>>(hw, dis, acc, NH);
    gcn_scatter  <<<cdiv(E * 8, B), B, 0, stream>>>(hw, src, dst, dis, acc, E);
    bias_relu    <<<cdiv(NH, B), B, 0, stream>>>(acc, gb2, out_h, NH);
    // Edge MLP
    const int nTileE = cdiv(E, 16);
    edge_mlp<<<cdiv(nTileE, 8), B, 0, stream>>>(out_h, src, dst, eatt, ew1t, eb1,
                                                ew2t, eb2, out_e, nTileE);
}
